// GPT_14903536517128
// MI455X (gfx1250) — compile-verified, matrix operand order checked by reading, not run on hardware
//
#include <hip/hip_runtime.h>
#include <hip/hip_bf16.h>
#include <math.h>

typedef __attribute__((ext_vector_type(16))) __bf16 v16bf;
typedef __attribute__((ext_vector_type(8)))  float  v8f;
typedef __attribute__((ext_vector_type(4)))  int    v4i;
typedef long long i64;

#define TT 1024
#define CC 1024
#define HH 16
#define DD 64
#define BB 2
#define VV 32000
#define NLAYER 6
#define MROWS (BB*TT)          // 2048
#define FF (4*CC)              // 4096

enum { OUT_ROW = 0, OUT_QKV = 1, OUT_VT = 2 };

// LDS tile: 128 rows x 32 bf16, row padded to 40 bf16 (80B) to spread banks.
#define LDS_STRIDE 40
#define LDS_SEC    (128 * LDS_STRIDE)   // 5120 bf16 = 10240B per tile buffer

// ---------------------------------------------------------------------------
// CDNA5 async global->LDS copy (16B per lane, ASYNCcnt-tracked) + wait
// ---------------------------------------------------------------------------
__device__ __forceinline__ void async_cp16(const __bf16* g, __bf16* l)
{
#if __has_builtin(__builtin_amdgcn_global_load_async_to_lds_b128)
    __builtin_amdgcn_global_load_async_to_lds_b128(
        (__attribute__((address_space(1))) v4i*)(void*)(const void*)g,
        (__attribute__((address_space(3))) v4i*)(void*)l, 0, 0);
#else
    unsigned loff = (unsigned)(unsigned long long)
        (__attribute__((address_space(3))) void*)l;
    asm volatile("global_load_async_to_lds_b128 %0, %1, off"
                 :: "v"(loff), "v"(g) : "memory");
#endif
}

template<int N>
__device__ __forceinline__ void wait_async()
{
#if __has_builtin(__builtin_amdgcn_s_wait_asynccnt)
    __builtin_amdgcn_s_wait_asynccnt((unsigned short)N);
#else
    if (N == 0) asm volatile("s_wait_asynccnt 0x0" ::: "memory");
    else        asm volatile("s_wait_asynccnt 0x4" ::: "memory");
#endif
}

// ---------------------------------------------------------------------------
// LDS-staged WMMA GEMM: OUT[m,n] = act(scale*sum_k A[m,k]*W[n,k] + bias) (+resid)
// A,W bf16; block 256 thr (8 waves) computes 128x128; wave = 32(M)x64(N)
// -> 8 v_wmma_f32_16x16x32_bf16 per BK=32 step, async double-buffered tiles.
// ---------------------------------------------------------------------------
template<bool BIAS, bool GELU, bool RESID, bool CAUSAL, bool OUT_BF16, int OUTL>
__global__ void __launch_bounds__(256)
gemm_lds(const __bf16* __restrict__ A, const __bf16* __restrict__ W,
         const float* __restrict__ bias, const float* __restrict__ resid,
         float* __restrict__ outF, __bf16* __restrict__ outB,
         int M, int N, int K, int lda, int ldw, int ldo,
         i64 sA, i64 sW, int outZdiv, i64 outS0, i64 outS1, float scale)
{
    __shared__ __bf16 smem[4 * LDS_SEC];          // [buf][A/B] double-buffered

    const int tid  = threadIdx.x;
    const int wave = tid >> 5;
    const int lane = tid & 31;
    const int hi   = lane >> 4;
    const int mr   = lane & 15;
    const int wm   = wave & 3;     // 4 waves over M (32 rows each)
    const int wn   = wave >> 2;    // 2 waves over N (64 cols each)

    const int z     = blockIdx.z;
    const int rowB0 = blockIdx.y * 128;
    const int colB0 = blockIdx.x * 128;

    A += (i64)z * sA;
    W += (i64)z * sW;
    const i64 outBase = (i64)(z / outZdiv) * outS0 + (i64)(z % outZdiv) * outS1;

    // causal fast path: tile entirely above the diagonal -> -inf, no GEMM
    if (CAUSAL && colB0 > rowB0 + 127) {
        for (int i = tid; i < 128 * 128; i += 256) {
            const int m = rowB0 + (i >> 7), n = colB0 + (i & 127);
            outF[outBase + (i64)m * ldo + n] = -__builtin_inff();
        }
        return;
    }

    // async tile copy: 512 chunks of 16B per tile, 2 per thread per matrix
    auto issue = [&](int buf, int kk) {
#pragma unroll
        for (int i = 0; i < 2; ++i) {
            const int c = tid + i * 256;           // 0..511
            const int row = c >> 2, q = c & 3;
            async_cp16(A + (i64)(rowB0 + row) * lda + kk + q * 8,
                       &smem[(buf * 2 + 0) * LDS_SEC + row * LDS_STRIDE + q * 8]);
        }
#pragma unroll
        for (int i = 0; i < 2; ++i) {
            const int c = tid + i * 256;
            const int row = c >> 2, q = c & 3;
            async_cp16(W + (i64)(colB0 + row) * ldw + kk + q * 8,
                       &smem[(buf * 2 + 1) * LDS_SEC + row * LDS_STRIDE + q * 8]);
        }
    };

    v8f acc[2][4] = {};
    issue(0, 0);
    const int KT = K >> 5;

    for (int kt = 0; kt < KT; ++kt) {
        const int cur = kt & 1;
        if (kt + 1 < KT) { issue(cur ^ 1, (kt + 1) * 32); wait_async<4>(); }
        else             { wait_async<0>(); }
        __syncthreads();

        const __bf16* As = &smem[(cur * 2 + 0) * LDS_SEC];
        const __bf16* Bs = &smem[(cur * 2 + 1) * LDS_SEC];

        v16bf a[2], b[4];
#pragma unroll
        for (int s = 0; s < 2; ++s) {
            const __bf16* ap = As + (wm * 32 + s * 16 + mr) * LDS_STRIDE + hi * 8;
#pragma unroll
            for (int j = 0; j < 8; ++j) { a[s][j] = ap[j]; a[s][8 + j] = ap[16 + j]; }
        }
#pragma unroll
        for (int t = 0; t < 4; ++t) {
            const __bf16* bp = Bs + (wn * 64 + t * 16 + mr) * LDS_STRIDE + hi * 8;
#pragma unroll
            for (int j = 0; j < 8; ++j) { b[t][j] = bp[j]; b[t][8 + j] = bp[16 + j]; }
        }
#pragma unroll
        for (int s = 0; s < 2; ++s)
#pragma unroll
            for (int t = 0; t < 4; ++t)
                acc[s][t] = __builtin_amdgcn_wmma_f32_16x16x32_bf16(
                    false, a[s], false, b[t], (short)0, acc[s][t], false, false);
        __syncthreads();
    }

    // epilogue: C/D layout lane(0..15)=N col, VGPR r = M row (+8 for hi lanes)
#pragma unroll
    for (int s = 0; s < 2; ++s)
#pragma unroll
        for (int t = 0; t < 4; ++t) {
            const int n = colB0 + wn * 64 + t * 16 + mr;
#pragma unroll
            for (int r = 0; r < 8; ++r) {
                const int m = rowB0 + wm * 32 + s * 16 + r + hi * 8;
                float v = acc[s][t][r] * scale;
                if (CAUSAL && (n > m)) v = -__builtin_inff();
                if (BIAS) v += bias[n];
                if (GELU) v = 0.5f * v * (1.0f + erff(v * 0.70710678118654752f));
                i64 off;
                if (OUTL == OUT_ROW) {
                    off = outBase + (i64)m * ldo + n;
                } else if (OUTL == OUT_QKV) {       // [B,H,T,D]
                    const int bb = m >> 10, tt = m & 1023, h = n >> 6, d = n & 63;
                    off = (((i64)bb * HH + h) * TT + tt) * DD + d;
                } else {                            // OUT_VT: [B,H,D,T]
                    const int bb = m >> 10, tt = m & 1023, h = n >> 6, d = n & 63;
                    off = (((i64)bb * HH + h) * DD + d) * TT + tt;
                }
                if (RESID) v += resid[off];
                if (OUT_BF16) outB[off] = (__bf16)v;
                else          outF[off] = v;
            }
        }
}

// ---------------------------------------------------------------------------
// Direct-load WMMA GEMM (for small-N att@V: N=64). A,W bf16 [.,K].
// ---------------------------------------------------------------------------
template<bool OUT_BF16, int OUTL>
__global__ void __launch_bounds__(128)
gemm_direct(const __bf16* __restrict__ A, const __bf16* __restrict__ W,
            float* __restrict__ outF, __bf16* __restrict__ outB,
            int M, int N, int K, int lda, int ldw, int ldo,
            i64 sA, i64 sW, int outZdiv, i64 outS0, i64 outS1)
{
    const int z    = blockIdx.z;
    const int wave = threadIdx.x >> 5;
    const int lane = threadIdx.x & 31;
    const int hi   = lane >> 4;
    const int mr   = lane & 15;

    const int row0 = blockIdx.y * 64 + wave * 16;
    const int col0 = blockIdx.x * 64;

    const __bf16* Ar = A + (i64)z * sA + (i64)(row0 + mr) * lda;
    const __bf16* Wz = W + (i64)z * sW;
    const i64 outBase = (i64)(z / outZdiv) * outS0 + (i64)(z % outZdiv) * outS1;

    v8f acc[4] = {};
    for (int kk = 0; kk < K; kk += 32) {
        v16bf a;
#pragma unroll
        for (int j = 0; j < 8; ++j) {
            a[j]     = Ar[kk + hi * 8 + j];
            a[8 + j] = Ar[kk + 16 + hi * 8 + j];
        }
#pragma unroll
        for (int t = 0; t < 4; ++t) {
            const __bf16* wr = Wz + (i64)(col0 + t * 16 + mr) * ldw + kk;
            v16bf b;
#pragma unroll
            for (int j = 0; j < 8; ++j) {
                b[j]     = wr[hi * 8 + j];
                b[8 + j] = wr[16 + hi * 8 + j];
            }
            acc[t] = __builtin_amdgcn_wmma_f32_16x16x32_bf16(
                false, a, false, b, (short)0, acc[t], false, false);
        }
    }
#pragma unroll
    for (int t = 0; t < 4; ++t) {
        const int n = col0 + t * 16 + mr;
#pragma unroll
        for (int r = 0; r < 8; ++r) {
            const int m = row0 + r + hi * 8;
            const float v = acc[t][r];
            const i64 off = outBase + (i64)m * ldo + n;
            if (OUT_BF16) outB[off] = (__bf16)v;
            else          outF[off] = v;
        }
    }
}

// ---------------------------------------------------------------------------
// fp32 -> bf16 weight conversion (grid covers n/4 threads, 4 elems each)
// ---------------------------------------------------------------------------
__global__ void __launch_bounds__(256)
cvt_bf16_k(const float* __restrict__ in, __bf16* __restrict__ out)
{
    const i64 base = ((i64)blockIdx.x * 256 + threadIdx.x) * 4;
#pragma unroll
    for (int j = 0; j < 4; ++j) out[base + j] = (__bf16)in[base + j];
}

// ---------------------------------------------------------------------------
__global__ void __launch_bounds__(256)
embed_k(const int* __restrict__ idx, const float* __restrict__ tok,
        const float* __restrict__ pos, const float* __restrict__ typ,
        float* __restrict__ x)
{
    const i64 i = (i64)blockIdx.x * 256 + threadIdx.x;
    const int c  = (int)(i & (CC - 1));
    const i64 bt = i >> 10;
    const int t  = (int)(bt & (TT - 1));
    x[i] = tok[(i64)idx[bt] * CC + c] + pos[(i64)t * CC + c] + typ[CC + c];
}

__global__ void __launch_bounds__(256)
layernorm_k(const float* __restrict__ x, const float* __restrict__ w,
            const float* __restrict__ b, __bf16* __restrict__ out)
{
    const int tid = threadIdx.x;
    const float* xr = x + (i64)blockIdx.x * CC;
    __shared__ float s1[256], s2[256];
    float a1 = 0.f, a2 = 0.f;
    for (int i = tid; i < CC; i += 256) { const float v = xr[i]; a1 += v; a2 += v * v; }
    s1[tid] = a1; s2[tid] = a2; __syncthreads();
    for (int st = 128; st > 0; st >>= 1) {
        if (tid < st) { s1[tid] += s1[tid + st]; s2[tid] += s2[tid + st]; }
        __syncthreads();
    }
    const float mean = s1[0] * (1.0f / CC);
    const float var  = s2[0] * (1.0f / CC) - mean * mean;
    const float inv  = rsqrtf(var + 1e-5f);
    __bf16* orow = out + (i64)blockIdx.x * CC;
    for (int i = tid; i < CC; i += 256)
        orow[i] = (__bf16)((xr[i] - mean) * inv * w[i] + b[i]);
}

__global__ void __launch_bounds__(256)
softmax_k(float* __restrict__ att, __bf16* __restrict__ attb)
{
    const int tid = threadIdx.x;
    float* r = att + (i64)blockIdx.x * TT;
    __shared__ float sh[256];
    float mx = -__builtin_inff();
    for (int j = 0; j < 4; ++j) mx = fmaxf(mx, r[tid + j * 256]);
    sh[tid] = mx; __syncthreads();
    for (int st = 128; st > 0; st >>= 1) {
        if (tid < st) sh[tid] = fmaxf(sh[tid], sh[tid + st]);
        __syncthreads();
    }
    mx = sh[0]; __syncthreads();
    float vals[4]; float se = 0.f;
    for (int j = 0; j < 4; ++j) { vals[j] = __expf(r[tid + j * 256] - mx); se += vals[j]; }
    sh[tid] = se; __syncthreads();
    for (int st = 128; st > 0; st >>= 1) {
        if (tid < st) sh[tid] += sh[tid + st];
        __syncthreads();
    }
    const float inv = 1.0f / sh[0];
    __bf16* rb = attb + (i64)blockIdx.x * TT;
    for (int j = 0; j < 4; ++j) {
        const float p = vals[j] * inv;
        r[tid + j * 256]  = p;
        rb[tid + j * 256] = (__bf16)p;
    }
}

__global__ void __launch_bounds__(256)
nll_k(const float* __restrict__ logits, const int* __restrict__ targets,
      float* __restrict__ nll, float* __restrict__ msk)
{
    const int tid = threadIdx.x;
    const float* lr = logits + (i64)blockIdx.x * VV;
    __shared__ float sh[256];
    float mx = -__builtin_inff();
    for (int i = tid; i < VV; i += 256) mx = fmaxf(mx, lr[i]);
    sh[tid] = mx; __syncthreads();
    for (int st = 128; st > 0; st >>= 1) {
        if (tid < st) sh[tid] = fmaxf(sh[tid], sh[tid + st]);
        __syncthreads();
    }
    mx = sh[0]; __syncthreads();
    float se = 0.f;
    for (int i = tid; i < VV; i += 256) se += __expf(lr[i] - mx);
    sh[tid] = se; __syncthreads();
    for (int st = 128; st > 0; st >>= 1) {
        if (tid < st) sh[tid] += sh[tid + st];
        __syncthreads();
    }
    if (tid == 0) {
        const int tg = targets[blockIdx.x];
        const float lp = lr[tg] - mx - logf(sh[0]);
        const float m  = (tg != 0) ? 1.0f : 0.0f;
        nll[blockIdx.x] = -lp * m;
        msk[blockIdx.x] = m;
    }
}

__global__ void __launch_bounds__(256)
loss_final_k(const float* __restrict__ nll, const float* __restrict__ msk,
             float* __restrict__ out)
{
    const int tid = threadIdx.x;
    __shared__ float s1[256], s2[256];
    float a = 0.f, c = 0.f;
    for (int i = tid; i < MROWS; i += 256) { a += nll[i]; c += msk[i]; }
    s1[tid] = a; s2[tid] = c; __syncthreads();
    for (int st = 128; st > 0; st >>= 1) {
        if (tid < st) { s1[tid] += s1[tid + st]; s2[tid] += s2[tid + st]; }
        __syncthreads();
    }
    if (tid == 0) out[0] = s1[0] / s2[0];
}

// ---------------------------------------------------------------------------
extern "C" void kernel_launch(void* const* d_in, const int* in_sizes, int n_in,
                              void* d_out, int out_size, void* d_ws, size_t ws_size,
                              hipStream_t stream)
{
    (void)in_sizes; (void)n_in; (void)out_size; (void)ws_size;

    const int* idx      = (const int*)d_in[0];
    const int* targets  = (const int*)d_in[1];
    const float* tok    = (const float*)d_in[2];
    const float* typ    = (const float*)d_in[3];
    const float* pos    = (const float*)d_in[4];
    const int WB = 5;
    const float* lnf_w  = (const float*)d_in[WB + NLAYER * 16 + 0];
    const float* lnf_b  = (const float*)d_in[WB + NLAYER * 16 + 1];
    const float* head_w = (const float*)d_in[WB + NLAYER * 16 + 2];

    float* outF = (float*)d_out;
    const i64 LOGITS = (i64)MROWS * VV;
    const i64 ATTSZ  = (i64)BB * HH * TT * TT;
    float* attnsOut  = outF + LOGITS + 1;

    char* wsp = (char*)d_ws;
    auto carve = [&](size_t bytes) { char* p = wsp; wsp += (bytes + 255) & ~(size_t)255; return p; };
    float*  x    = (float*) carve((size_t)MROWS * CC * 4);
    __bf16* h    = (__bf16*)carve((size_t)MROWS * CC * 2);
    __bf16* q    = (__bf16*)carve((size_t)MROWS * CC * 2);   // [B,H,T,D]
    __bf16* k    = (__bf16*)carve((size_t)MROWS * CC * 2);   // [B,H,T,D]
    __bf16* vt   = (__bf16*)carve((size_t)MROWS * CC * 2);   // [B,H,D,T]
    __bf16* y    = (__bf16*)carve((size_t)MROWS * CC * 2);   // attn out [B,T,C]
    __bf16* f1   = (__bf16*)carve((size_t)MROWS * FF * 2);
    __bf16* attb = (__bf16*)carve((size_t)ATTSZ * 2);
    float*  nll  = (float*) carve((size_t)MROWS * 4);
    float*  msk  = (float*) carve((size_t)MROWS * 4);
    // bf16 weight copies
    __bf16* wq[NLAYER]; __bf16* wk[NLAYER]; __bf16* wv[NLAYER];
    __bf16* wp[NLAYER]; __bf16* w1[NLAYER]; __bf16* w2[NLAYER];
    for (int l = 0; l < NLAYER; ++l) {
        wq[l] = (__bf16*)carve((size_t)CC * CC * 2);
        wk[l] = (__bf16*)carve((size_t)CC * CC * 2);
        wv[l] = (__bf16*)carve((size_t)CC * CC * 2);
        wp[l] = (__bf16*)carve((size_t)CC * CC * 2);
        w1[l] = (__bf16*)carve((size_t)FF * CC * 2);
        w2[l] = (__bf16*)carve((size_t)CC * FF * 2);
    }
    __bf16* whead = (__bf16*)carve((size_t)VV * CC * 2);

    const dim3 blk128(128), blk256(256);
    const int CVT_CC = (CC * CC) / 1024, CVT_FC = (FF * CC) / 1024;

    // ---- convert all GEMM weights to bf16 once ----
    for (int l = 0; l < NLAYER; ++l) {
        cvt_bf16_k<<<dim3(CVT_CC), blk256, 0, stream>>>((const float*)d_in[WB + l*16 + 4],  wq[l]);
        cvt_bf16_k<<<dim3(CVT_CC), blk256, 0, stream>>>((const float*)d_in[WB + l*16 + 6],  wk[l]);
        cvt_bf16_k<<<dim3(CVT_CC), blk256, 0, stream>>>((const float*)d_in[WB + l*16 + 8],  wv[l]);
        cvt_bf16_k<<<dim3(CVT_CC), blk256, 0, stream>>>((const float*)d_in[WB + l*16 + 10], wp[l]);
        cvt_bf16_k<<<dim3(CVT_FC), blk256, 0, stream>>>((const float*)d_in[WB + l*16 + 12], w1[l]);
        cvt_bf16_k<<<dim3(CVT_FC), blk256, 0, stream>>>((const float*)d_in[WB + l*16 + 14], w2[l]);
    }
    cvt_bf16_k<<<dim3((VV * CC) / 1024), blk256, 0, stream>>>(head_w, whead);

    // ---- embeddings ----
    embed_k<<<dim3((MROWS * CC) / 256), blk256, 0, stream>>>(idx, tok, pos, typ, x);

    for (int l = 0; l < NLAYER; ++l) {
        const float* ln1_w = (const float*)d_in[WB + l*16 + 0];
        const float* ln1_b = (const float*)d_in[WB + l*16 + 1];
        const float* ln2_w = (const float*)d_in[WB + l*16 + 2];
        const float* ln2_b = (const float*)d_in[WB + l*16 + 3];
        const float* q_b   = (const float*)d_in[WB + l*16 + 5];
        const float* k_b   = (const float*)d_in[WB + l*16 + 7];
        const float* v_b   = (const float*)d_in[WB + l*16 + 9];
        const float* p_b   = (const float*)d_in[WB + l*16 + 11];
        const float* f1_b  = (const float*)d_in[WB + l*16 + 13];
        const float* f2_b  = (const float*)d_in[WB + l*16 + 15];
        float* attL = attnsOut + (i64)l * ATTSZ;

        layernorm_k<<<dim3(MROWS), blk256, 0, stream>>>(x, ln1_w, ln1_b, h);

        gemm_lds<true, false, false, false, true, OUT_QKV>
            <<<dim3(CC/128, MROWS/128, 1), blk256, 0, stream>>>(
                h, wq[l], q_b, nullptr, nullptr, q,
                MROWS, CC, CC, CC, CC, CC, 0, 0, 1, 0, 0, 1.0f);
        gemm_lds<true, false, false, false, true, OUT_QKV>
            <<<dim3(CC/128, MROWS/128, 1), blk256, 0, stream>>>(
                h, wk[l], k_b, nullptr, nullptr, k,
                MROWS, CC, CC, CC, CC, CC, 0, 0, 1, 0, 0, 1.0f);
        gemm_lds<true, false, false, false, true, OUT_VT>
            <<<dim3(CC/128, MROWS/128, 1), blk256, 0, stream>>>(
                h, wv[l], v_b, nullptr, nullptr, vt,
                MROWS, CC, CC, CC, CC, CC, 0, 0, 1, 0, 0, 1.0f);

        // scores: q @ k^T / 8 with causal mask, fp32 into d_out attn slab
        gemm_lds<false, false, false, true, false, OUT_ROW>
            <<<dim3(TT/128, TT/128, BB*HH), blk256, 0, stream>>>(
                q, k, nullptr, nullptr, attL, nullptr,
                TT, TT, DD, DD, DD, TT,
                (i64)TT*DD, (i64)TT*DD, 1, (i64)TT*TT, 0, 0.125f);

        softmax_k<<<dim3(BB*HH*TT), blk256, 0, stream>>>(attL, attb);

        // y = att @ v (v stored transposed), gather to [B,T,C]
        gemm_direct<true, OUT_ROW>
            <<<dim3(DD/64, TT/64, BB*HH), blk128, 0, stream>>>(
                attb, vt, nullptr, y,
                TT, DD, TT, TT, TT, CC,
                (i64)TT*TT, (i64)DD*TT, HH, (i64)TT*CC, (i64)DD);

        gemm_lds<true, false, true, false, false, OUT_ROW>
            <<<dim3(CC/128, MROWS/128, 1), blk256, 0, stream>>>(
                y, wp[l], p_b, x, x, nullptr,
                MROWS, CC, CC, CC, CC, CC, 0, 0, 1, 0, 0, 1.0f);

        layernorm_k<<<dim3(MROWS), blk256, 0, stream>>>(x, ln2_w, ln2_b, h);

        gemm_lds<true, true, false, false, true, OUT_ROW>
            <<<dim3(FF/128, MROWS/128, 1), blk256, 0, stream>>>(
                h, w1[l], f1_b, nullptr, nullptr, f1,
                MROWS, FF, CC, CC, CC, FF, 0, 0, 1, 0, 0, 1.0f);

        gemm_lds<true, false, true, false, false, OUT_ROW>
            <<<dim3(CC/128, MROWS/128, 1), blk256, 0, stream>>>(
                f1, w2[l], f2_b, x, x, nullptr,
                MROWS, CC, FF, FF, FF, CC, 0, 0, 1, 0, 0, 1.0f);
    }

    layernorm_k<<<dim3(MROWS), blk256, 0, stream>>>(x, lnf_w, lnf_b, h);

    // head: logits = h @ head_w^T  (N = 32000 = 250*128)
    gemm_lds<false, false, false, false, false, OUT_ROW>
        <<<dim3(VV/128, MROWS/128, 1), blk256, 0, stream>>>(
            h, whead, nullptr, nullptr, outF, nullptr,
            MROWS, VV, CC, CC, CC, VV, 0, 0, 1, 0, 0, 1.0f);

    nll_k<<<dim3(MROWS), blk256, 0, stream>>>(outF, targets, nll, msk);
    loss_final_k<<<dim3(1), blk256, 0, stream>>>(nll, msk, outF + LOGITS);
}
